// SeqAttention_56856777064588
// MI455X (gfx1250) — matrix operand order repaired and checked by reading, not compiled, not run on hardware
//
#include <hip/hip_runtime.h>

// MI455X / gfx1250, wave32. Sliding-window relative attention (flash style).
// B=128, M=512, L=1024, H=64, keylen N=M+L=1536.

typedef __attribute__((ext_vector_type(16))) _Float16 v16h;
typedef __attribute__((ext_vector_type(8)))  float    v8f;

#define WMMA_F16(A, B, C) \
  __builtin_amdgcn_wmma_f32_16x16x32_f16(false, (A), false, (B), (short)0, (C), false, false)

static __device__ __forceinline__ v8f zero8() {
  v8f r;
  #pragma unroll
  for (int j = 0; j < 8; ++j) r[j] = 0.0f;
  return r;
}

// 16 contiguous f32 -> 16 f16 lanes of a WMMA-B fragment (compiler emits b128 loads)
static __device__ __forceinline__ v16h cvt16_contig(const float* __restrict__ p) {
  v16h r;
  #pragma unroll
  for (int u = 0; u < 16; ++u) r[u] = (_Float16)p[u];
  return r;
}

__global__ __launch_bounds__(32)
void relattn_fwd(const float* __restrict__ Q, const float* __restrict__ K,
                 const float* __restrict__ V, const float* __restrict__ PE,
                 float* __restrict__ O) {
  constexpr int M = 512, N = 1536, L = 1024, H = 64;

  // Per-wave LDS: 4-slot ring of 16x16 relative-position score chunks + P staging.
  __shared__ float    posbuf[4][16 * 16];
  __shared__ _Float16 pbuf[16 * 32];

  const int lane  = threadIdx.x;      // 0..31
  const int col   = lane & 15;        // N index inside a 16-wide tile
  const int hi    = lane >> 4;        // half-wave select
  const int bidx  = blockIdx.x >> 5;  // batch*head
  const int m0    = (blockIdx.x & 31) << 4;

  const float* Qb = Q + (size_t)bidx * M * H;
  const float* Kb = K + (size_t)bidx * N * H;
  const float* Vb = V + (size_t)bidx * N * H;

  // ---- Q tile as WMMA-A operand (M=lane%16; contiguous k order chosen here: K = u + (hi?16:0)) ----
  v16h aq[2];
  {
    const float* qrow = Qb + (size_t)(m0 + col) * H;
    const int off = hi ? 16 : 0;
    #pragma unroll
    for (int kc = 0; kc < 2; ++kc) {
      #pragma unroll
      for (int u = 0; u < 8; ++u) aq[kc][u]     = (_Float16)qrow[kc * 32 + off + u];
      #pragma unroll
      for (int u = 0; u < 8; ++u) aq[kc][u + 8] = (_Float16)qrow[kc * 32 + off + 8 + u];
    }
  }

  const int Mbase = hi << 3;  // C-layout row base for this half-wave
  float mrow[8], lsum[8];
  #pragma unroll
  for (int r = 0; r < 8; ++r) { mrow[r] = -3.0e38f; lsum[r] = 0.0f; }
  v8f o[4];
  #pragma unroll
  for (int t = 0; t < 4; ++t) o[t] = zero8();

  // 33 blocks of 32 absolute keys cover the band [m0, m0+15+1023].
  for (int i = 0; i <= 32; ++i) {
    const int kb = m0 + (i << 5);

    if (i < 32) {  // CDNA5 global_prefetch_b8 on next K/V tiles
      __builtin_prefetch(Kb + (size_t)(kb + 32 + col) * H, 0, 1);
      __builtin_prefetch(Vb + (size_t)(kb + 32 + col) * H, 0, 1);
    }

    // ---- relative-position chunks tt = 2i, 2i+1: P_pos[m][l] = Q·PE[:,l] via WMMA ----
    #pragma unroll
    for (int q = 0; q < 2; ++q) {
      const int tt = 2 * i + q;           // wave-uniform (EXEC stays all-ones)
      if (tt < 64) {
        const int lcol = (tt << 4) + col; // l < 1024, in bounds
        v8f pc = zero8();
        #pragma unroll
        for (int kc = 0; kc < 2; ++kc) {
          v16h bpe;  // B[k=h][n=l], k = kc*32 + hi*16 + u
          #pragma unroll
          for (int u = 0; u < 16; ++u)
            bpe[u] = (_Float16)PE[(size_t)(kc * 32 + (hi << 4) + u) * L + lcol];
          pc = WMMA_F16(aq[kc], bpe, pc);
        }
        #pragma unroll
        for (int r = 0; r < 8; ++r)
          posbuf[tt & 3][(Mbase + r) * 16 + col] = pc[r];
      }
    }

    // ---- content scores S = Q K^T for keys [kb, kb+32) ----
    v8f c0 = zero8(), c1 = zero8();
    {
      const int key0 = min(kb + col, N - 1);       // edge rows are band-masked below
      const int key1 = min(kb + 16 + col, N - 1);
      #pragma unroll
      for (int kc = 0; kc < 2; ++kc) {
        v16h bk0 = cvt16_contig(Kb + (size_t)key0 * H + kc * 32 + (hi << 4));
        c0 = WMMA_F16(aq[kc], bk0, c0);
        v16h bk1 = cvt16_contig(Kb + (size_t)key1 * H + kc * 32 + (hi << 4));
        c1 = WMMA_F16(aq[kc], bk1, c1);
      }
    }

    // ---- add skewed position term (LDS gather = in-register unskew), mask band ----
    float s0[8], s1[8];
    const int n0 = kb + col, n1 = n0 + 16;
    #pragma unroll
    for (int r = 0; r < 8; ++r) {
      const int m  = m0 + Mbase + r;
      const int l0 = n0 - m, l1 = n1 - m;
      s0[r] = (l0 >= 0 && l0 < L)
            ? 0.125f * (c0[r] + posbuf[(l0 >> 4) & 3][(Mbase + r) * 16 + (l0 & 15)])
            : -3.0e38f;
      s1[r] = (l1 >= 0 && l1 < L)
            ? 0.125f * (c1[r] + posbuf[(l1 >> 4) & 3][(Mbase + r) * 16 + (l1 & 15)])
            : -3.0e38f;
    }

    // ---- online softmax (rows live across 16 lanes; xor masks 1..8 stay in-half) ----
    float alpha[8];
    #pragma unroll
    for (int r = 0; r < 8; ++r) {
      float v = fmaxf(s0[r], s1[r]);
      #pragma unroll
      for (int d = 1; d <= 8; d <<= 1) v = fmaxf(v, __shfl_xor(v, d, 32));
      const float mn = fmaxf(mrow[r], v);
      alpha[r] = __expf(mrow[r] - mn);
      mrow[r]  = mn;
      const float p0 = __expf(s0[r] - mn);
      const float p1 = __expf(s1[r] - mn);
      s0[r] = p0; s1[r] = p1;
      float ps = p0 + p1;
      #pragma unroll
      for (int d = 1; d <= 8; d <<= 1) ps += __shfl_xor(ps, d, 32);
      lsum[r] = lsum[r] * alpha[r] + ps;
    }
    #pragma unroll
    for (int t = 0; t < 4; ++t) {
      #pragma unroll
      for (int r = 0; r < 8; ++r) o[t][r] *= alpha[r];
    }

    // ---- P: C-layout f32 -> LDS f16 -> A-layout fragment ----
    #pragma unroll
    for (int r = 0; r < 8; ++r) {
      pbuf[(Mbase + r) * 32 + col]      = (_Float16)s0[r];
      pbuf[(Mbase + r) * 32 + 16 + col] = (_Float16)s1[r];
    }
    asm volatile("s_wait_dscnt 0" ::: "memory");
    v16h pa;
    {
      const _Float16* pr = pbuf + col * 32 + (hi ? 16 : 0);
      #pragma unroll
      for (int u = 0; u < 8; ++u) pa[u]     = pr[u];
      #pragma unroll
      for (int u = 0; u < 8; ++u) pa[u + 8] = pr[8 + u];
    }

    // ---- O += P · V  (B[k=key][n=h], k = hi*16 + u; masked rows have P==0) ----
    #pragma unroll
    for (int t = 0; t < 4; ++t) {
      v16h bv;
      #pragma unroll
      for (int u = 0; u < 16; ++u) {
        const int key = min(kb + (hi << 4) + u, N - 1);
        bv[u] = (_Float16)Vb[(size_t)key * H + t * 16 + col];
      }
      o[t] = WMMA_F16(pa, bv, o[t]);
    }
  }

  // ---- epilogue: normalize and store (f32 out) ----
  #pragma unroll
  for (int r = 0; r < 8; ++r) lsum[r] = 1.0f / lsum[r];
  float* Ob = O + ((size_t)bidx * M + m0) * H;
  #pragma unroll
  for (int t = 0; t < 4; ++t) {
    #pragma unroll
    for (int r = 0; r < 8; ++r)
      Ob[(size_t)(Mbase + r) * H + t * 16 + col] = o[t][r] * lsum[r];
  }
}

extern "C" void kernel_launch(void* const* d_in, const int* in_sizes, int n_in,
                              void* d_out, int out_size, void* d_ws, size_t ws_size,
                              hipStream_t stream) {
  (void)n_in; (void)out_size; (void)d_ws; (void)ws_size;
  const float* Q  = (const float*)d_in[0];  // B x 512 x 64
  const float* K  = (const float*)d_in[1];  // B x 1536 x 64
  const float* V  = (const float*)d_in[2];  // B x 1536 x 64
  const float* PE = (const float*)d_in[3];  // 1 x 64 x 1024
  float* out = (float*)d_out;               // B x 512 x 64

  const int B = in_sizes[0] / (512 * 64);   // 128
  dim3 grid(B * 32);                        // 32 query tiles of 16 per batch-head
  dim3 block(32);                           // one wave32 per workgroup
  relattn_fwd<<<grid, block, 0, stream>>>(Q, K, V, PE, out);
}
